// GCNEdgePredictorWithEmbeddings_80762565034207
// MI455X (gfx1250) — matrix operand order repaired and checked by reading, not compiled, not run on hardware
//
#include <hip/hip_runtime.h>
#include <hip/hip_bf16.h>

// ---------------------------------------------------------------------------
// GCN edge predictor for MI455X (gfx1250, wave32, WMMA).
// Heavy path: two 8192x8192x{256,128} aggregations with
// v_wmma_f32_16x16x32_bf16 (f32 accumulate), double-buffered LDS:
//   A operand: attn (f32) relu/deg^-1/2-scaled, v_perm pair-packed to bf16,
//              immediate-offset global loads off one running base pointer.
//   B operand: pre-converted/pre-transposed bf16 [C][N], copied into LDS with
//              global_load_async_to_lds_b128 (ASYNCcnt) at immediate offsets.
// One barrier per K-step: next step's staging overlaps current step's WMMAs.
// ---------------------------------------------------------------------------

typedef __attribute__((ext_vector_type(16))) __bf16 v16bf;
typedef __attribute__((ext_vector_type(8)))  float  v8f;

union Frag32 { uint4 q[2]; v16bf v; };   // 32 bytes = 8 VGPRs = one bf16 A/B fragment

template<int BYTEOFF>
__device__ __forceinline__ void asyncB128(unsigned lds, unsigned long long ga) {
    // INST_OFFSET is added to BOTH the LDS and global address (ISA 15.18 async),
    // which matches our 1:1 chunk layout.
    asm volatile("global_load_async_to_lds_b128 %0, %1, off offset:%2"
                 :: "v"(lds), "v"(ga), "i"(BYTEOFF) : "memory");
}

// --------------------------- small helpers ---------------------------------

__global__ void zero_kernel(float* __restrict__ p, int n) {
    int i = blockIdx.x * blockDim.x + threadIdx.x;
    if (i < n) p[i] = 0.0f;
}

__global__ void colsum_kernel(const float* __restrict__ attn, float* __restrict__ deg, int n) {
    const int col  = blockIdx.x * blockDim.x + threadIdx.x;
    const int rows = n / gridDim.y;
    const int r0   = blockIdx.y * rows;
    float s = 0.0f;
    for (int r = r0; r < r0 + rows; ++r)
        s += fmaxf(attn[(size_t)r * n + col], 0.0f);
    atomicAdd(&deg[col], s);
}

__global__ void dis_kernel(const float* __restrict__ deg, float* __restrict__ dis, int n) {
    int i = blockIdx.x * blockDim.x + threadIdx.x;
    if (i < n) { float d = deg[i]; dis[i] = (d > 0.0f) ? rsqrtf(d) : 0.0f; }
}

// out[row][c] = sum_k In[row][k] * W[k][c]   (f32; bias fused later)
template<int K, int C>
__global__ __launch_bounds__(256) void rowgemm_kernel(const float* __restrict__ In,
                                                      const float* __restrict__ W,
                                                      float* __restrict__ out) {
    __shared__ float xs[K];
    const int row = blockIdx.x;
    for (int k = threadIdx.x; k < K; k += blockDim.x)
        xs[k] = In[(size_t)row * K + k];
    __syncthreads();
    const int c = threadIdx.x;          // blockDim.x == C
    float acc = 0.0f;
#pragma unroll 8
    for (int k = 0; k < K; ++k)
        acc += xs[k] * W[k * C + c];
    out[(size_t)row * C + c] = acc;
}

// f32 [R][C] row-major -> bf16 [C][R] row-major (round-half-up), 64x64 LDS tiles
__global__ __launch_bounds__(256) void tconv_kernel(const float* __restrict__ In,
                                                    unsigned short* __restrict__ OutT,
                                                    int R, int C) {
    constexpr int TS = 64, STR = 80;
    __shared__ __align__(16) unsigned short tile[TS * STR];
    const int r0 = blockIdx.x * TS;
    const int c0 = blockIdx.y * TS;
    const int t  = threadIdx.x;
#pragma unroll
    for (int p = 0; p < 4; ++p) {
        const int q  = t + p * 256;
        const int r  = q >> 4;
        const int c4 = (q & 15) * 4;
        const float4 v = *(const float4*)(In + (size_t)(r0 + r) * C + c0 + c4);
        tile[(c4 + 0) * STR + r] = (unsigned short)((__float_as_uint(v.x) + 0x8000u) >> 16);
        tile[(c4 + 1) * STR + r] = (unsigned short)((__float_as_uint(v.y) + 0x8000u) >> 16);
        tile[(c4 + 2) * STR + r] = (unsigned short)((__float_as_uint(v.z) + 0x8000u) >> 16);
        tile[(c4 + 3) * STR + r] = (unsigned short)((__float_as_uint(v.w) + 0x8000u) >> 16);
    }
    __syncthreads();
#pragma unroll
    for (int p = 0; p < 2; ++p) {
        const int q  = t + p * 256;
        const int cc = q >> 3;
        const int ch = q & 7;
        const uint4 w = *(const uint4*)(tile + cc * STR + ch * 8);
        *(uint4*)(OutT + (size_t)(c0 + cc) * R + r0 + ch * 8) = w;
    }
}

// --------------------------- WMMA aggregation ------------------------------
// out[i][c] = epilogue( dis[i] * sum_j (dis[j]*relu(attn[j][i])) * Min[j][c] )
// 256 threads = 8 waves; wave tile 32(i) x 64(c) = 2x4 WMMA frags.
// CDIM=256 -> 4 c-waves x 2 i-waves (ITILE=64); CDIM=128 -> 2x4 (ITILE=128).
template<int N, int CDIM, bool RELU>
__global__ __launch_bounds__(256) void gcn_agg_kernel(const float* __restrict__ attn,
                                                      const float* __restrict__ dis,
                                                      const unsigned short* __restrict__ MinT, // bf16 [CDIM][N]
                                                      const float* __restrict__ bias,
                                                      float* __restrict__ out) {
    constexpr int CW     = CDIM / 64;
    constexpr int IW     = 8 / CW;
    constexpr int ITILE  = 32 * IW;
    constexpr int LSTR   = 40;              // ushorts/row: 80B, 16B-aligned, bank-spread
    constexpr int PPP    = 256 / ITILE;     // j-pairs per pass (4 or 2)
    constexpr int REPS   = 16 / PPP;        // 4 or 8
    constexpr int PARTS  = 256 / CDIM;      // 1 or 2 threads per B row
    constexpr size_t ASTEP = (size_t)32 * N; // floats per K-step

    __shared__ __align__(16) unsigned short lsA[2][ITILE * LSTR]; // [i_local][jj]
    __shared__ __align__(16) unsigned short lsB[2][CDIM  * LSTR]; // [c][jj]

    const int t     = threadIdx.x;
    const int iBase = blockIdx.x * ITILE;
    const int wave  = t >> 5, lane = t & 31;
    const int wI    = wave / CW, wC = wave % CW;
    const int laneN = lane & 15, laneH = lane >> 4;

    v8f acc[2][4] = {};

    // ---- A staging state: one running base pointer, constant rep offsets ----
    const int aI = t % ITILE, aP = t / ITILE;
    const float* aPtr = attn + (size_t)(2 * aP) * N + iBase + aI;   // += ASTEP / step
    const float* dPtr = dis + 2 * aP;                                // += 32 / step
    unsigned short* const aDst0 = &lsA[0][aI * LSTR + 2 * aP];
    unsigned short* const aDst1 = &lsA[1][aI * LSTR + 2 * aP];

    // ---- B async state: one row (or half-row) per thread ----
    const int bC = t % CDIM, bPart = t / CDIM;        // bPart in [0,PARTS)
    unsigned long long bGa =
        (unsigned long long)(size_t)(MinT + (size_t)bC * N + bPart * 16);  // += 64B / step
    const unsigned bLds0 = (unsigned)(size_t)(const void*)&lsB[0][bC * LSTR + bPart * 16];
    const unsigned bLds1 = (unsigned)(size_t)(const void*)&lsB[1][bC * LSTR + bPart * 16];

    auto issueB = [&](int buf) {
        const unsigned l = buf ? bLds1 : bLds0;
        asyncB128<0>(l, bGa);
        asyncB128<16>(l, bGa);
        if constexpr (PARTS == 1) {
            asyncB128<32>(l, bGa);
            asyncB128<48>(l, bGa);
        }
        bGa += 64;
    };

    auto stageA = [&](int buf) {
        unsigned short* dst = buf ? aDst1 : aDst0;
#pragma unroll
        for (int r = 0; r < REPS; ++r) {
            const float2 s2 = *(const float2*)(dPtr + 2 * PPP * r);
            const float  a0 = aPtr[(size_t)(2 * PPP * r) * N];
            const float  a1 = aPtr[(size_t)(2 * PPP * r) * N + N];
            const unsigned u0 = __float_as_uint(fmaxf(a0, 0.0f) * s2.x) + 0x8000u;
            const unsigned u1 = __float_as_uint(fmaxf(a1, 0.0f) * s2.y) + 0x8000u;
            *(unsigned*)(dst + 2 * PPP * r) = __builtin_amdgcn_perm(u1, u0, 0x07060302u);
            // global_prefetch_b8 two K-steps ahead of use
            __builtin_prefetch(aPtr + (size_t)(2 * PPP * r) * N + 2 * ASTEP, 0, 1);
        }
        aPtr += ASTEP;
        dPtr += 32;
    };

    // Prologue: fill buffer 0 for j0 = 0.
    issueB(0);
    stageA(0);

    int cur = 0;
#pragma unroll 1
    for (int j0 = 0; j0 < N; j0 += 32) {
        asm volatile("s_wait_asynccnt 0" ::: "memory");
        __syncthreads();
        const int nxt = cur ^ 1;
        if (j0 + 32 < N) {          // overlap next staging with current WMMAs
            issueB(nxt);
            stageA(nxt);
        }

        // A fragments (16x32 bf16): lane m<16 -> K0..15, m>=16 -> K16..31
        Frag32 af[2];
#pragma unroll
        for (int s = 0; s < 2; ++s) {
            const uint4* p = (const uint4*)(&lsA[cur][(wI * 32 + s * 16 + laneN) * LSTR]);
            af[s].q[0] = p[laneH * 2];
            af[s].q[1] = p[laneH * 2 + 1];
        }
        // B fragments (32x16 bf16): lane n<16 -> K0..15, n>=16 -> K16..31 of column n
        Frag32 bfr[4];
#pragma unroll
        for (int q = 0; q < 4; ++q) {
            const uint4* p = (const uint4*)(&lsB[cur][(wC * 64 + q * 16 + laneN) * LSTR]);
            bfr[q].q[0] = p[laneH * 2];
            bfr[q].q[1] = p[laneH * 2 + 1];
        }
#pragma unroll
        for (int s = 0; s < 2; ++s)
#pragma unroll
            for (int q = 0; q < 4; ++q)
                acc[s][q] = __builtin_amdgcn_wmma_f32_16x16x32_bf16(
                    false, af[s].v, false, bfr[q].v, (short)0, acc[s][q], false, false);
        cur = nxt;
    }

    // Epilogue: *dis[i] + bias[c] (+relu). C/D layout: lane<16 -> M=r, lane>=16 -> M=8+r.
#pragma unroll
    for (int s = 0; s < 2; ++s) {
        const int i0 = iBase + wI * 32 + s * 16 + laneH * 8;
        float dv[8];
#pragma unroll
        for (int r = 0; r < 8; ++r) dv[r] = dis[i0 + r];
#pragma unroll
        for (int q = 0; q < 4; ++q) {
            const int   c  = wC * 64 + q * 16 + laneN;
            const float bc = bias[c];
#pragma unroll
            for (int r = 0; r < 8; ++r) {
                float v = acc[s][q][r] * dv[r] + bc;
                if (RELU) v = fmaxf(v, 0.0f);
                out[(size_t)(i0 + r) * CDIM + c] = v;
            }
        }
    }
}

// --------------------------- edge decode -----------------------------------
__global__ __launch_bounds__(256) void decode_kernel(const float* __restrict__ z,
                                                     const int* __restrict__ eidx,
                                                     float* __restrict__ out, int npred) {
    const int lane = threadIdx.x & 31;
    const int gw   = blockIdx.x * (blockDim.x >> 5) + (threadIdx.x >> 5);
    const int nw   = gridDim.x * (blockDim.x >> 5);
    for (int e = gw; e < npred; e += nw) {
        const int s = eidx[e];
        const int d = eidx[npred + e];
        const float4 a = ((const float4*)(z + (size_t)s * 128))[lane];
        const float4 b = ((const float4*)(z + (size_t)d * 128))[lane];
        float acc = a.x * b.x + a.y * b.y + a.z * b.z + a.w * b.w;
#pragma unroll
        for (int off = 16; off > 0; off >>= 1)
            acc += __shfl_xor(acc, off, 32);
        if (lane == 0) out[e] = acc;
    }
}

// --------------------------- launch ----------------------------------------

extern "C" void kernel_launch(void* const* d_in, const int* in_sizes, int n_in,
                              void* d_out, int out_size, void* d_ws, size_t ws_size,
                              hipStream_t stream) {
    (void)n_in; (void)out_size; (void)ws_size;
    // inputs: node_indices(unused, identity), attn, edge_label_index, emb, W1, b1, W2, b2
    const float* attn = (const float*)d_in[1];
    const int*   eidx = (const int*)d_in[2];
    const float* emb  = (const float*)d_in[3];
    const float* W1   = (const float*)d_in[4];
    const float* b1   = (const float*)d_in[5];
    const float* W2   = (const float*)d_in[6];
    const float* b2   = (const float*)d_in[7];

    constexpr int NN = 8192;              // fixed problem size (template-folded offsets)
    const int n     = in_sizes[0];        // == NN
    const int npred = in_sizes[2] / 2;    // 2,000,000
    constexpr int ED = 256, HID = 256, OUTD = 128;
    constexpr size_t MB = 1024 * 1024;

    // ws: dis(32K) | deg(32K) | bufA 8MB (XW f32 -> hW2 f32 + hW2T bf16)
    //     | bufB 8MB (h f32 -> z f32) | XWT bf16 4MB          (~20.1 MB)
    char*  ws   = (char*)d_ws;
    float* dis  = (float*)(ws);
    float* deg  = (float*)(ws + 32768);
    float* bufA = (float*)(ws + 65536);
    float* bufB = (float*)(ws + 65536 + 8 * MB);
    unsigned short* XWT  = (unsigned short*)(ws + 65536 + 16 * MB);
    unsigned short* hW2T = (unsigned short*)(ws + 65536 + 4 * MB);  // 2nd half of bufA region
    float* scores = (float*)d_out;

    zero_kernel<<<(n + 255) / 256, 256, 0, stream>>>(deg, n);
    colsum_kernel<<<dim3(n / 256, 16), 256, 0, stream>>>(attn, deg, n);
    dis_kernel<<<(n + 255) / 256, 256, 0, stream>>>(deg, dis, n);

    rowgemm_kernel<ED, HID><<<n, HID, 0, stream>>>(emb, W1, bufA);                 // XW f32
    tconv_kernel<<<dim3(n / 64, HID / 64), 256, 0, stream>>>(bufA, XWT, n, HID);   // -> bf16 [HID][n]
    gcn_agg_kernel<NN, HID, true><<<NN / 64, 256, 0, stream>>>(attn, dis, XWT, b1, bufB);   // h

    rowgemm_kernel<HID, OUTD><<<n, OUTD, 0, stream>>>(bufB, W2, bufA);             // hW2 f32 (4MB)
    tconv_kernel<<<dim3(n / 64, OUTD / 64), 256, 0, stream>>>(bufA, hW2T, n, OUTD);// -> bf16 [OUTD][n]
    gcn_agg_kernel<NN, OUTD, false><<<NN / 128, 256, 0, stream>>>(attn, dis, hW2T, b2, bufB); // z

    decode_kernel<<<4096, 256, 0, stream>>>(bufB, eidx, scores, npred);
}
